// SelfAttention_74448963109280
// MI455X (gfx1250) — matrix operand order repaired and checked by reading, not compiled, not run on hardware
//
#include <hip/hip_runtime.h>

typedef __bf16 bf16;
typedef __attribute__((ext_vector_type(8)))  float v8f;
typedef __attribute__((ext_vector_type(8)))  bf16  v8bf;
typedef __attribute__((ext_vector_type(16))) bf16  v16bf;

static constexpr int BATCH = 4, SEQ = 2048, DM = 1024, NH = 16, DH = 64;
static constexpr int ROWS = BATCH * SEQ;   // 8192
static constexpr int F3   = 3 * DM;        // 3072
static constexpr int LDP  = 72;            // padded LDS row stride (144 B, 16B-aligned, bank-spread)

// ---------- helpers ----------

__device__ inline bf16 f2bf(float x) {
  union { float f; unsigned u; } a; a.f = x;
  unsigned r = a.u + 0x7FFFu + ((a.u >> 16) & 1u);   // round-to-nearest-even
  union { unsigned short s; bf16 b; } o; o.s = (unsigned short)(r >> 16);
  return o.b;
}

__device__ inline v8f wmma_bf16(v16bf a, v16bf b, v8f c) {
  return __builtin_amdgcn_wmma_f32_16x16x32_bf16(
      /*neg_a=*/false, a, /*neg_b=*/false, b,
      /*c_mod=*/(short)0, c, /*reuse_a=*/false, /*reuse_b=*/false);
}

// A-matrix 16x32 lane layout: half h holds K = off+{0..7} and off+{16..23}, off = k0+8*h
__device__ inline v16bf lds_read_a(const bf16* rowp, int off) {
  union { v16bf v; v8bf h[2]; } u;
  u.h[0] = *(const v8bf*)(rowp + off);
  u.h[1] = *(const v8bf*)(rowp + off + 16);
  return u.v;
}
// B-matrix 32x16 lane layout: half h holds K = off+{0..7} and off+{16..23}, off = k0+8*h
__device__ inline v16bf lds_read_b(const bf16* rowp, int off) {
  union { v16bf v; v8bf h[2]; } u;
  u.h[0] = *(const v8bf*)(rowp + off);
  u.h[1] = *(const v8bf*)(rowp + off + 16);
  return u.v;
}

// ---- gfx1250 async copy: global -> LDS, tracked by ASYNCcnt (no VGPR staging) ----
__device__ inline unsigned lds_off32(const void* p) {
  // generic pointer to __shared__: low 32 bits are the LDS byte address
  return (unsigned)(unsigned long long)p;
}
__device__ inline void async_ld_b128(unsigned lds, const void* gaddr) {
  asm volatile("global_load_async_to_lds_b128 %0, %1, off"
               :: "v"(lds), "v"(gaddr) : "memory");
}
__device__ inline void wait_async0() {
  asm volatile("s_wait_asynccnt 0x0" ::: "memory");
}

// ---------- kernel 0: f32 -> bf16 ----------

__global__ void cvt_f32_bf16(const float* __restrict__ src, bf16* __restrict__ dst, int n) {
  int i = blockIdx.x * blockDim.x + threadIdx.x;
  if (i < n) dst[i] = f2bf(src[i]);
}

// ---------- shared GEMM mainloop: C[128x128] = A[128xK] * W[128xK]^T ----------
// 256 threads = 8 waves; wave (wm,wn) computes rows wm*32..+31, cols wn*64..+63.
// Double-buffered LDS, async global->LDS fills, one barrier per 64-wide K chunk.

__device__ inline void gemm_mainloop(const bf16* __restrict__ A, const bf16* __restrict__ W,
                                     int blockM, int blockN,
                                     bf16* As, bf16* Bs, v8f acc[2][4]) {
  const int tid  = threadIdx.x;
  const int w    = tid >> 5, lane = tid & 31;
  const int half = lane >> 4, ln = lane & 15;
  const int wm   = w & 3, wn = w >> 2;

  auto issue = [&](int p, int kc) {
    bf16* ab = As + p * 128 * LDP;
    bf16* bb = Bs + p * 128 * LDP;
    #pragma unroll
    for (int i = 0; i < 4; ++i) {              // 128x64 tile: 4 x 16B per thread per array
      int e = (tid + i * 256) * 8;
      int row = e >> 6, col = e & 63;
      async_ld_b128(lds_off32(&ab[row * LDP + col]), &A[(size_t)(blockM + row) * DM + kc + col]);
      async_ld_b128(lds_off32(&bb[row * LDP + col]), &W[(size_t)(blockN + row) * DM + kc + col]);
    }
  };

  issue(0, 0);
  for (int it = 0; it < DM / 64; ++it) {
    const int p = it & 1;
    wait_async0();          // my fills of buffer p have landed
    __syncthreads();        // everyone's fills landed; everyone done reading buffer p^1
    if (it + 1 < DM / 64) issue(p ^ 1, (it + 1) * 64);   // overlap next fill with compute
    const bf16* as = As + p * 128 * LDP;
    const bf16* bs = Bs + p * 128 * LDP;
    #pragma unroll
    for (int kk = 0; kk < 2; ++kk) {           // two k=32 WMMA steps per chunk
      const int k0 = kk * 32;
      v16bf av[2], bv[4];
      #pragma unroll
      for (int mt = 0; mt < 2; ++mt)
        av[mt] = lds_read_a(&as[(wm * 32 + mt * 16 + ln) * LDP], k0 + 8 * half);
      #pragma unroll
      for (int nt = 0; nt < 4; ++nt)
        bv[nt] = lds_read_b(&bs[(wn * 64 + nt * 16 + ln) * LDP], k0 + 8 * half);
      #pragma unroll
      for (int mt = 0; mt < 2; ++mt)
        #pragma unroll
        for (int nt = 0; nt < 4; ++nt)
          acc[mt][nt] = wmma_bf16(av[mt], bv[nt], acc[mt][nt]);
    }
  }
}

// ---------- kernel 1: QKV projection, scatter to (B,H,T,D) bf16; Q pre-scaled 1/sqrt(Dh) ----------

__global__ __launch_bounds__(256) void gemm_qkv(const bf16* __restrict__ A, const bf16* __restrict__ W,
                                                bf16* __restrict__ Qb, bf16* __restrict__ Kb,
                                                bf16* __restrict__ Vb) {
  __shared__ bf16 As[2 * 128 * LDP];
  __shared__ bf16 Bs[2 * 128 * LDP];
  const int tid = threadIdx.x;
  const int w = tid >> 5, lane = tid & 31, half = lane >> 4, ln = lane & 15;
  const int wm = w & 3, wn = w >> 2;
  const int blockM = blockIdx.x * 128, blockN = blockIdx.y * 128;
  v8f acc[2][4] = {};
  gemm_mainloop(A, W, blockM, blockN, As, Bs, acc);
  #pragma unroll
  for (int mt = 0; mt < 2; ++mt)
    #pragma unroll
    for (int nt = 0; nt < 4; ++nt)
      #pragma unroll
      for (int r = 0; r < 8; ++r) {
        int row = blockM + wm * 32 + mt * 16 + r + 8 * half;   // global token row
        int f   = blockN + wn * 64 + nt * 16 + ln;             // output feature in [0,3C)
        float v = acc[mt][nt][r];
        int type = f >> 10, c = f & 1023;
        int hh = c >> 6, d = c & 63;
        int bb = row >> 11, t = row & 2047;
        size_t di = ((size_t)(bb * NH + hh) * SEQ + t) * DH + d;
        if (type == 0)      Qb[di] = f2bf(v * 0.125f);   // 1/sqrt(64)
        else if (type == 1) Kb[di] = f2bf(v);
        else                Vb[di] = f2bf(v);
      }
}

// ---------- kernel 2: causal flash attention, 64-query tile per block (4 waves) ----------

__global__ __launch_bounds__(128) void flash_attn(const bf16* __restrict__ Qg,
                                                  const bf16* __restrict__ Kg,
                                                  const bf16* __restrict__ Vg,
                                                  bf16* __restrict__ Ob) {
  __shared__ bf16 Qs[64 * LDP];    // query tile (rows x d)
  __shared__ bf16 Ks[64 * LDP];    // key tile (keys x d)
  __shared__ bf16 Vts[64 * LDP];   // V transposed (d x keys)
  __shared__ bf16 Ps[64 * LDP];    // P staging (rows x keys)
  const int tid = threadIdx.x;
  const int w = tid >> 5, lane = tid & 31, half = lane >> 4, ln = lane & 15;
  const int qtile = blockIdx.x, bh = blockIdx.y;
  const int q0 = qtile * 64;
  const size_t base = (size_t)bh * SEQ * DH;

  // Q tile: async global -> LDS
  #pragma unroll
  for (int i = 0; i < 4; ++i) {
    int e = (tid + i * 128) * 8;
    int r = e >> 6, c = e & 63;
    async_ld_b128(lds_off32(&Qs[r * LDP + c]), &Qg[base + (size_t)(q0 + r) * DH + c]);
  }
  wait_async0();
  __syncthreads();
  v16bf qa0 = lds_read_a(&Qs[(w * 16 + ln) * LDP],  0 + 8 * half);
  v16bf qa1 = lds_read_a(&Qs[(w * 16 + ln) * LDP], 32 + 8 * half);

  v8f O[4] = {};
  float m[8], l[8];
  #pragma unroll
  for (int r = 0; r < 8; ++r) { m[r] = -1e30f; l[r] = 0.f; }

  for (int j = 0; j <= qtile; ++j) {
    // K tile: async -> LDS; V tile: register transpose into Vts (async can't transpose)
    #pragma unroll
    for (int i = 0; i < 4; ++i) {
      int e = (tid + i * 128) * 8;
      int r = e >> 6, c = e & 63;
      async_ld_b128(lds_off32(&Ks[r * LDP + c]), &Kg[base + (size_t)(j * 64 + r) * DH + c]);
      v8bf vv = *(const v8bf*)&Vg[base + (size_t)(j * 64 + r) * DH + c];
      #pragma unroll
      for (int x = 0; x < 8; ++x) Vts[(c + x) * LDP + r] = vv[x];
    }
    wait_async0();
    __syncthreads();

    // S = Q * K^T  (16x64 per wave)
    v8f S[4] = {};
    #pragma unroll
    for (int nt = 0; nt < 4; ++nt) {
      const bf16* krow = &Ks[(nt * 16 + ln) * LDP];
      S[nt] = wmma_bf16(qa0, lds_read_b(krow,  0 + 8 * half), S[nt]);
      S[nt] = wmma_bf16(qa1, lds_read_b(krow, 32 + 8 * half), S[nt]);
    }
    if (j == qtile) {                       // causal mask on diagonal tile only
      #pragma unroll
      for (int nt = 0; nt < 4; ++nt)
        #pragma unroll
        for (int r = 0; r < 8; ++r)
          if ((nt * 16 + ln) > (w * 16 + r + 8 * half)) S[nt][r] = -1e30f;
    }

    // online softmax
    float tm[8], rs[8];
    #pragma unroll
    for (int r = 0; r < 8; ++r) {
      float v0 = fmaxf(fmaxf(S[0][r], S[1][r]), fmaxf(S[2][r], S[3][r]));
      v0 = fmaxf(v0, __shfl_xor(v0, 1));
      v0 = fmaxf(v0, __shfl_xor(v0, 2));
      v0 = fmaxf(v0, __shfl_xor(v0, 4));
      v0 = fmaxf(v0, __shfl_xor(v0, 8));
      tm[r] = v0;
    }
    #pragma unroll
    for (int r = 0; r < 8; ++r) {
      float newm = fmaxf(m[r], tm[r]);
      float sc = __expf(m[r] - newm);
      m[r] = newm;
      l[r] *= sc;
      #pragma unroll
      for (int dt = 0; dt < 4; ++dt) O[dt][r] *= sc;
      rs[r] = 0.f;
    }
    bf16* prow0 = &Ps[(w * 16 + 8 * half) * LDP];
    #pragma unroll
    for (int nt = 0; nt < 4; ++nt)
      #pragma unroll
      for (int r = 0; r < 8; ++r) {
        float p = __expf(S[nt][r] - m[r]);
        rs[r] += p;
        prow0[r * LDP + nt * 16 + ln] = f2bf(p);
      }
    #pragma unroll
    for (int r = 0; r < 8; ++r) {
      float v0 = rs[r];
      v0 += __shfl_xor(v0, 1);
      v0 += __shfl_xor(v0, 2);
      v0 += __shfl_xor(v0, 4);
      v0 += __shfl_xor(v0, 8);
      l[r] += v0;
    }

    // O += P * V  (wave-private Ps region; in-wave DS ops are ordered)
    v16bf pa0 = lds_read_a(&Ps[(w * 16 + ln) * LDP],  0 + 8 * half);
    v16bf pa1 = lds_read_a(&Ps[(w * 16 + ln) * LDP], 32 + 8 * half);
    #pragma unroll
    for (int dt = 0; dt < 4; ++dt) {
      const bf16* vrow = &Vts[(dt * 16 + ln) * LDP];
      O[dt] = wmma_bf16(pa0, lds_read_b(vrow,  0 + 8 * half), O[dt]);
      O[dt] = wmma_bf16(pa1, lds_read_b(vrow, 32 + 8 * half), O[dt]);
    }
    __syncthreads();
  }

  // normalize, store to (B,T,C) bf16 for the output projection
  const int b = bh >> 4, h = bh & 15;
  #pragma unroll
  for (int r = 0; r < 8; ++r) {
    float inv = 1.0f / l[r];
    int t = q0 + w * 16 + r + 8 * half;
    size_t orow = ((size_t)b * SEQ + t) * DM + h * DH;
    #pragma unroll
    for (int dt = 0; dt < 4; ++dt)
      Ob[orow + dt * 16 + ln] = f2bf(O[dt][r] * inv);
  }
}

// ---------- kernel 3: output projection, f32 result ----------

__global__ __launch_bounds__(256) void gemm_out(const bf16* __restrict__ A, const bf16* __restrict__ W,
                                                float* __restrict__ C) {
  __shared__ bf16 As[2 * 128 * LDP];
  __shared__ bf16 Bs[2 * 128 * LDP];
  const int tid = threadIdx.x;
  const int w = tid >> 5, lane = tid & 31, half = lane >> 4, ln = lane & 15;
  const int wm = w & 3, wn = w >> 2;
  const int blockM = blockIdx.x * 128, blockN = blockIdx.y * 128;
  v8f acc[2][4] = {};
  gemm_mainloop(A, W, blockM, blockN, As, Bs, acc);
  #pragma unroll
  for (int mt = 0; mt < 2; ++mt)
    #pragma unroll
    for (int nt = 0; nt < 4; ++nt)
      #pragma unroll
      for (int r = 0; r < 8; ++r) {
        int row = blockM + wm * 32 + mt * 16 + r + 8 * half;
        int col = blockN + wn * 64 + nt * 16 + ln;
        C[(size_t)row * DM + col] = acc[mt][nt][r];
      }
}

// ---------- launcher ----------

extern "C" void kernel_launch(void* const* d_in, const int* in_sizes, int n_in,
                              void* d_out, int out_size, void* d_ws, size_t ws_size,
                              hipStream_t stream) {
  (void)in_sizes; (void)n_in; (void)out_size; (void)ws_size;
  const float* x    = (const float*)d_in[0];
  const float* Wqkv = (const float*)d_in[1];
  const float* Wo   = (const float*)d_in[2];
  float* out = (float*)d_out;

  char* ws = (char*)d_ws;
  size_t off = 0;
  bf16* xb    = (bf16*)(ws + off); off += (size_t)ROWS * DM * 2;
  bf16* wqkvb = (bf16*)(ws + off); off += (size_t)F3 * DM * 2;
  bf16* wob   = (bf16*)(ws + off); off += (size_t)DM * DM * 2;
  bf16* Qb    = (bf16*)(ws + off); off += (size_t)ROWS * DM * 2;
  bf16* Kb    = (bf16*)(ws + off); off += (size_t)ROWS * DM * 2;
  bf16* Vb    = (bf16*)(ws + off); off += (size_t)ROWS * DM * 2;
  bf16* attnb = xb;   // x_bf16 is dead after gemm_qkv; reuse for attention output

  cvt_f32_bf16<<<(ROWS * DM + 255) / 256, 256, 0, stream>>>(x, xb, ROWS * DM);
  cvt_f32_bf16<<<(F3 * DM + 255) / 256, 256, 0, stream>>>(Wqkv, wqkvb, F3 * DM);
  cvt_f32_bf16<<<(DM * DM + 255) / 256, 256, 0, stream>>>(Wo, wob, DM * DM);

  gemm_qkv<<<dim3(ROWS / 128, F3 / 128), 256, 0, stream>>>(xb, wqkvb, Qb, Kb, Vb);
  flash_attn<<<dim3(SEQ / 64, BATCH * NH), 128, 0, stream>>>(Qb, Kb, Vb, attnb);
  gemm_out<<<dim3(ROWS / 128, DM / 128), 256, 0, stream>>>(attnb, wob, out);
}
